// PointNet_v1_6579889897759
// MI455X (gfx1250) — matrix operand order repaired and checked by reading, not compiled, not run on hardware
//
#include <hip/hip_runtime.h>
#include <hip/hip_bf16.h>
#include <math.h>

typedef __bf16 bf16_t;
typedef __bf16 bfv16 __attribute__((ext_vector_type(16)));
typedef __bf16 bfv8  __attribute__((ext_vector_type(8)));
typedef float  fv8   __attribute__((ext_vector_type(8)));

#define NPTS 8192
#define NB   8
#define BN   65536   // NB * NPTS

union ABu { bfv16 v; bfv8 h[2]; };

// ---------------- weight convert f32 -> bf16 ----------------
__global__ void cvt_f32_bf16(const float* __restrict__ src, bf16_t* __restrict__ dst, int n) {
  int i = blockIdx.x * blockDim.x + threadIdx.x;
  if (i < n) dst[i] = (bf16_t)src[i];
}

// ---------------- zero accumulators (ws is poisoned) ----------------
__global__ void init_accums(float* __restrict__ gsum, unsigned long long* __restrict__ pool) {
  int i = blockIdx.x * blockDim.x + threadIdx.x;
  if (i < 4096) gsum[i] = 0.f;
  if (i < 8192) pool[i] = 0ull;
}

// ---------------- layer0 (7->64, VALU) + group id ----------------
__global__ __launch_bounds__(256) void layer0_gid(
    const float* __restrict__ pts, const float* __restrict__ w0,
    const float* __restrict__ b0, bf16_t* __restrict__ h0, int* __restrict__ gid) {
  __shared__ float sw[64 * 7];
  __shared__ float sb[64];
  for (int i = threadIdx.x; i < 64 * 7; i += blockDim.x) sw[i] = w0[i];
  if (threadIdx.x < 64) sb[threadIdx.x] = b0[threadIdx.x];
  __syncthreads();
  int p = blockIdx.x * blockDim.x + threadIdx.x;   // 0..65535
  int b = p >> 13, n = p & (NPTS - 1);
  const float* xp = pts + (size_t)b * 7 * NPTS + n;
  float x[7];
#pragma unroll
  for (int c = 0; c < 7; ++c) x[c] = xp[(size_t)c * NPTS];
  // argmax over channels 3..6, first-max tie-break
  int g = 0; float bv = x[3];
#pragma unroll
  for (int c = 1; c < 4; ++c) if (x[3 + c] > bv) { bv = x[3 + c]; g = c; }
  gid[p] = g;
  bf16_t o[64];
#pragma unroll
  for (int oo = 0; oo < 64; ++oo) {
    float acc = sb[oo];
#pragma unroll
    for (int c = 0; c < 7; ++c) acc = fmaf(sw[oo * 7 + c], x[c], acc);
    o[oo] = (bf16_t)(acc > 0.f ? acc : 0.f);
  }
  bfv8* dst = (bfv8*)(h0 + (size_t)p * 64);
#pragma unroll
  for (int i = 0; i < 8; ++i) dst[i] = ((bfv8*)o)[i];
}

// ---------------- WMMA GEMM: Y[p,m] = relu(W[m,:] . X[p,:] + bias[m]) ----------------
// X point-major [BN,K] bf16, W row-major [M,K] bf16, Y point-major [BN,M] bf16.
// Each wave holds MTW m-tiles of A-fragments resident and issues MTW independent
// v_wmma_f32_16x16x32_bf16 per B-fragment load (register blocking in M cuts the
// redundant L2 reads of X by MTW and gives MTW-way WMMA ILP per wave).
template <int M, int K, int PT_ITER, int MTW>
__global__ __launch_bounds__(256) void gemm_relu(
    const bf16_t* __restrict__ X, const bf16_t* __restrict__ W,
    const float* __restrict__ bias, bf16_t* __restrict__ Y) {
  constexpr int MGROUPS = M / (16 * MTW);
  constexpr int WPB = (MGROUPS < 8) ? MGROUPS : 8;   // waves per block
  const int wave = threadIdx.x >> 5;
  const int lane = threadIdx.x & 31;
  const int hi   = lane >> 4;                        // 0/1 half-wave
  const int l15  = lane & 15;
  const int mt0  = (blockIdx.y * WPB + wave) * MTW;  // first m-tile of this wave

  constexpr int KF = K / 32;
  // A fragments (ISA 16-bit A layout: lanes0-15 K={0..7,16..23}, lanes16-31 K={8..15,24..31})
  bfv16 a[MTW][KF];
#pragma unroll
  for (int j = 0; j < MTW; ++j) {
    const bf16_t* wp = W + (size_t)((mt0 + j) * 16 + l15) * K + hi * 8;
#pragma unroll
    for (int kf = 0; kf < KF; ++kf) {
      ABu u;
      u.h[0] = *(const bfv8*)(wp + kf * 32);
      u.h[1] = *(const bfv8*)(wp + kf * 32 + 16);
      a[j][kf] = u.v;
    }
  }

#pragma unroll
  for (int it = 0; it < PT_ITER; ++it) {
    const int p = (blockIdx.x * PT_ITER + it) * 16 + l15;   // point (column)
    const bf16_t* xp = X + (size_t)p * K + hi * 8;          // B layout: same K order as A
    fv8 c[MTW] = {};
#pragma unroll
    for (int kf = 0; kf < KF; ++kf) {
      ABu ub;
      ub.h[0] = *(const bfv8*)(xp + kf * 32);               // two 16B loads
      ub.h[1] = *(const bfv8*)(xp + kf * 32 + 16);
      bfv16 bm = ub.v;
#pragma unroll
      for (int j = 0; j < MTW; ++j)
        c[j] = __builtin_amdgcn_wmma_f32_16x16x32_bf16(false, a[j][kf], false, bm,
                                                       (short)0, c[j], false, false);
    }
#pragma unroll
    for (int j = 0; j < MTW; ++j) {
      const int mo = (mt0 + j) * 16 + hi * 8;               // first of 8 output rows
      bfv8 o;
#pragma unroll
      for (int r = 0; r < 8; ++r) {
        float v = c[j][r] + bias[mo + r];
        o[r] = (bf16_t)(v > 0.f ? v : 0.f);
      }
      *(bfv8*)(Y + (size_t)p * M + mo) = o;                 // 16B contiguous store
    }
  }
}

// ---------------- per-group sums over u ----------------
__global__ __launch_bounds__(1024) void group_sums(
    const bf16_t* __restrict__ U, const int* __restrict__ gid,
    float* __restrict__ gsum) {
  __shared__ int sg[1024];
  const int d = threadIdx.x;
  const int b = blockIdx.y;
  const int pbase = b * NPTS + blockIdx.x * 1024;
  sg[d] = gid[pbase + d];
  __syncthreads();
  float s0 = 0, s1 = 0, s2 = 0, s3 = 0;
  for (int i = 0; i < 1024; ++i) {
    float v = (float)U[(size_t)(pbase + i) * 1024 + d];     // 2KB coalesced row
    int g = sg[i];
    s0 += (g == 0) ? v : 0.f;
    s1 += (g == 1) ? v : 0.f;
    s2 += (g == 2) ? v : 0.f;
    s3 += (g == 3) ? v : 0.f;
  }
  atomicAdd(&gsum[d], s0);
  atomicAdd(&gsum[1024 + d], s1);
  atomicAdd(&gsum[2048 + d], s2);
  atomicAdd(&gsum[3072 + d], s3);
}

// ---------------- excitation MLP (single block) ----------------
__global__ __launch_bounds__(1024) void excite(
    const float* __restrict__ gsum, const int* __restrict__ gid,
    const float* __restrict__ we1, const float* __restrict__ be1,
    const float* __restrict__ we2, const float* __restrict__ be2,
    float* __restrict__ act) {
  __shared__ float means[4096];
  __shared__ float e1s[512];
  __shared__ int cnt[4];
  const int t = threadIdx.x;
  if (t < 4) cnt[t] = 0;
  __syncthreads();
  int c0 = 0, c1 = 0, c2 = 0, c3 = 0;
  for (int p = t; p < BN; p += 1024) {
    int g = gid[p];
    c0 += (g == 0); c1 += (g == 1); c2 += (g == 2); c3 += (g == 3);
  }
  atomicAdd(&cnt[0], c0); atomicAdd(&cnt[1], c1);
  atomicAdd(&cnt[2], c2); atomicAdd(&cnt[3], c3);
  __syncthreads();
  for (int i = t; i < 4096; i += 1024) {
    float cn = (float)cnt[i >> 10];
    means[i] = gsum[i] / (cn > 1.f ? cn : 1.f);
  }
  __syncthreads();
  if (t < 512) {
    int g = t >> 7, o = t & 127;
    float acc = be1[o];
    for (int k = 0; k < 1024; ++k) acc = fmaf(means[g * 1024 + k], we1[o * 1024 + k], acc);
    e1s[t] = acc > 0.f ? acc : 0.f;
  }
  __syncthreads();
  for (int i = t; i < 4096; i += 1024) {
    int g = i >> 10, d = i & 1023;
    float acc = be2[d];
#pragma unroll 8
    for (int k = 0; k < 128; ++k) acc = fmaf(e1s[g * 128 + k], we2[d * 128 + k], acc);
    act[i] = 1.f / (1.f + __expf(-acc));
  }
}

// ---------------- scaled max pool with argmax (packed u64 atomic max) ----------------
__global__ __launch_bounds__(1024) void pool_max(
    const bf16_t* __restrict__ U, const int* __restrict__ gid,
    const float* __restrict__ act, unsigned long long* __restrict__ pool) {
  __shared__ int sg[256];
  const int d = threadIdx.x;
  const int b = blockIdx.y;
  const int n0 = blockIdx.x * 256;
  const int pbase = b * NPTS + n0;
  if (d < 256) sg[d] = gid[pbase + d];
  __syncthreads();
  const float a0 = act[d], a1 = act[1024 + d], a2 = act[2048 + d], a3 = act[3072 + d];
  float best = -1.f; int bestn = n0;
  for (int i = 0; i < 256; ++i) {
    float u = (float)U[(size_t)(pbase + i) * 1024 + d];
    int g = sg[i];
    float s = (g == 0) ? a0 : ((g == 1) ? a1 : ((g == 2) ? a2 : a3));
    float v = u * s;
    if (v > best) { best = v; bestn = n0 + i; }            // strict > keeps first index
  }
  // v >= 0 always (relu * sigmoid): f32 bits are order preserving; inverted
  // index makes ties pick the smallest n under max().
  unsigned long long key =
      ((unsigned long long)__float_as_uint(best) << 32) |
      (unsigned long long)(unsigned)(NPTS - 1 - bestn);
  atomicMax(&pool[b * 1024 + d], key);
}

__global__ void decode_pool(const unsigned long long* __restrict__ pool,
                            float* __restrict__ gf, float* __restrict__ idx_out) {
  int i = blockIdx.x * blockDim.x + threadIdx.x;
  if (i >= 8192) return;
  unsigned long long k = pool[i];
  gf[i] = __uint_as_float((unsigned)(k >> 32));
  idx_out[i] = (float)(NPTS - 1 - (int)(unsigned)(k & 0xffffffffu));
}

// ---------------- head MLP 1024->512->256 (single block) ----------------
__global__ __launch_bounds__(512) void head_mlp(
    const float* __restrict__ gf, const float* __restrict__ wg1,
    const float* __restrict__ bg1, const float* __restrict__ wg2,
    const float* __restrict__ bg2, float* __restrict__ out) {
  __shared__ float g_s[8 * 512];
  const int t = threadIdx.x;
  {
    float acc[8];
    float bb = bg1[t];
#pragma unroll
    for (int b = 0; b < 8; ++b) acc[b] = bb;
    for (int k = 0; k < 1024; ++k) {
      float w = wg1[t * 1024 + k];
#pragma unroll
      for (int b = 0; b < 8; ++b) acc[b] = fmaf(gf[b * 1024 + k], w, acc[b]);
    }
#pragma unroll
    for (int b = 0; b < 8; ++b) g_s[b * 512 + t] = acc[b] > 0.f ? acc[b] : 0.f;
  }
  __syncthreads();
  if (t < 256) {
    float acc[8];
    float bb = bg2[t];
#pragma unroll
    for (int b = 0; b < 8; ++b) acc[b] = bb;
    for (int k = 0; k < 512; ++k) {
      float w = wg2[t * 512 + k];
#pragma unroll
      for (int b = 0; b < 8; ++b) acc[b] = fmaf(g_s[b * 512 + k], w, acc[b]);
    }
#pragma unroll
    for (int b = 0; b < 8; ++b) out[b * 256 + t] = acc[b] > 0.f ? acc[b] : 0.f;
  }
}

extern "C" void kernel_launch(void* const* d_in, const int* in_sizes, int n_in,
                              void* d_out, int out_size, void* d_ws, size_t ws_size,
                              hipStream_t stream) {
  (void)in_sizes; (void)n_in; (void)out_size; (void)ws_size;
  const float* pts = (const float*)d_in[0];
  const float* w0 = (const float*)d_in[1];  const float* b0 = (const float*)d_in[2];
  const float* w1 = (const float*)d_in[3];  const float* b1 = (const float*)d_in[4];
  const float* w2 = (const float*)d_in[5];  const float* b2 = (const float*)d_in[6];
  const float* w3 = (const float*)d_in[7];  const float* b3 = (const float*)d_in[8];
  const float* w4 = (const float*)d_in[9];  const float* b4 = (const float*)d_in[10];
  const float* we1 = (const float*)d_in[11]; const float* be1 = (const float*)d_in[12];
  const float* we2 = (const float*)d_in[13]; const float* be2 = (const float*)d_in[14];
  const float* wg1 = (const float*)d_in[15]; const float* bg1 = (const float*)d_in[16];
  const float* wg2 = (const float*)d_in[17]; const float* bg2 = (const float*)d_in[18];

  char* ws = (char*)d_ws;
  size_t off = 0;
  auto alloc = [&](size_t bytes) {
    size_t o = off; off = (off + bytes + 255) & ~(size_t)255; return o;
  };
  bf16_t* wb1 = (bf16_t*)(ws + alloc(64 * 64 * 2));
  bf16_t* wb2 = (bf16_t*)(ws + alloc(64 * 64 * 2));
  bf16_t* wb3 = (bf16_t*)(ws + alloc(128 * 64 * 2));
  bf16_t* wb4 = (bf16_t*)(ws + alloc(1024 * 128 * 2));
  int*    gid  = (int*)(ws + alloc((size_t)BN * 4));
  float*  gsum = (float*)(ws + alloc(4096 * 4));
  float*  act  = (float*)(ws + alloc(4096 * 4));
  unsigned long long* pool = (unsigned long long*)(ws + alloc(8192 * 8));
  float*  gf   = (float*)(ws + alloc(8192 * 4));
  bf16_t* hA = (bf16_t*)(ws + alloc((size_t)BN * 64 * 2));    //  8 MB ping
  bf16_t* hB = (bf16_t*)(ws + alloc((size_t)BN * 128 * 2));   // 16 MB pong
  bf16_t* U  = (bf16_t*)(ws + alloc((size_t)BN * 1024 * 2));  // 128 MB

  float* out_f = (float*)d_out;            // [0,2048): out ; [2048,10240): indices

  // weight converts + accumulator init
  cvt_f32_bf16<<<(4096 + 255) / 256, 256, 0, stream>>>(w1, wb1, 4096);
  cvt_f32_bf16<<<(4096 + 255) / 256, 256, 0, stream>>>(w2, wb2, 4096);
  cvt_f32_bf16<<<(8192 + 255) / 256, 256, 0, stream>>>(w3, wb3, 8192);
  cvt_f32_bf16<<<(131072 + 255) / 256, 256, 0, stream>>>(w4, wb4, 131072);
  init_accums<<<32, 256, 0, stream>>>(gsum, pool);

  // per-point MLP chain
  layer0_gid<<<BN / 256, 256, 0, stream>>>(pts, w0, b0, hA, gid);
  gemm_relu<64, 64, 4, 1><<<dim3(1024, 1), 128, 0, stream>>>(hA, wb1, b1, hB);
  gemm_relu<64, 64, 4, 1><<<dim3(1024, 1), 128, 0, stream>>>(hB, wb2, b2, hA);
  gemm_relu<128, 64, 4, 1><<<dim3(1024, 1), 256, 0, stream>>>(hA, wb3, b3, hB);
  gemm_relu<1024, 128, 4, 4><<<dim3(1024, 2), 256, 0, stream>>>(hB, wb4, b4, U);

  // group means -> excitation -> scaled max pool
  group_sums<<<dim3(8, 8), 1024, 0, stream>>>(U, gid, gsum);
  excite<<<1, 1024, 0, stream>>>(gsum, gid, we1, be1, we2, be2, act);
  pool_max<<<dim3(32, 8), 1024, 0, stream>>>(U, gid, act, pool);
  decode_pool<<<32, 256, 0, stream>>>(pool, gf, out_f + 2048);

  // head
  head_mlp<<<1, 512, 0, stream>>>(gf, wg1, bg1, wg2, bg2, out_f);
}
